// DiffusionModel_56822417326086
// MI455X (gfx1250) — matrix operand order repaired and checked by reading, not compiled, not run on hardware
//
#include <hip/hip_runtime.h>

typedef __attribute__((ext_vector_type(16))) __bf16 v16bf;
typedef __attribute__((ext_vector_type(8)))  float  v8f;

#define B_   4
#define T_   2048
#define D_   512
#define H_   8
#define DH_  64
#define SCALE_ 0.125f
#define NEGBIG -3.0e38f

// ------------------------------------------------------------ fragment math
// A-matrix (16x32 bf16): lane = m + 16*((k%16)>=8), element idx = 8*(k/16) + (k%8)
static __device__ __forceinline__ int a_lane(int m, int k) {
  return m + (((k & 15) >= 8) ? 16 : 0);
}
static __device__ __forceinline__ int a_idx(int k) {
  return ((k >> 4) << 3) | (k & 7);
}
// B-matrix (32x16 bf16): lane = n + 16*((k%16)>=8), element idx = 8*(k/16) + (k%8)
static __device__ __forceinline__ int b_lane(int n, int k) {
  return n + (((k & 15) >= 8) ? 16 : 0);
}
static __device__ __forceinline__ int b_idx(int k) { return ((k >> 4) << 3) | (k & 7); }

static __device__ __forceinline__ v8f wmma_bf16(v16bf a, v16bf b, v8f c) {
  return __builtin_amdgcn_wmma_f32_16x16x32_bf16(false, a, false, b, (short)0, c,
                                                 false, false);
}

// Fragment-packed load: base points at a [32][16] bf16 block (32B/lane, aligned)
static __device__ __forceinline__ v16bf ld_frag(const __bf16* __restrict__ base) {
  return *(const v16bf*)(base + (size_t)(threadIdx.x & 31) * 16);
}

static __device__ __forceinline__ float rmax16(float v) {
#pragma unroll
  for (int s = 1; s < 16; s <<= 1) v = fmaxf(v, __shfl_xor(v, s, 32));
  return v;
}
static __device__ __forceinline__ float rsum16(float v) {
#pragma unroll
  for (int s = 1; s < 16; s <<= 1) v += __shfl_xor(v, s, 32);
  return v;
}

// FRAG = one packed fragment = 32*16 = 512 bf16 elements (1 KB)
#define FRAG 512

// ------------------------------------------------------------ pack kernels

// x [8192,512] f32 -> A-frag-packed bf16 [mt=512][kt=16][lane][16].
// One thread builds one fragment-lane: two 8-float contiguous runs of one row,
// 4x b128 loads + one 32B vector store, coalesced both sides.
__global__ __launch_bounds__(256) void pack_x_kernel(
    const float* __restrict__ x, __bf16* __restrict__ Xp) {
  const int idx = blockIdx.x * 256 + threadIdx.x;  // 512*16*32 threads
  const int lane = idx & 31;
  const int kt = (idx >> 5) & 15;
  const int mt = idx >> 9;
  const int m = lane & 15;
  const int kb = (lane & 16) ? 8 : 0;
  const float* src = x + (size_t)(mt * 16 + m) * D_ + kt * 32 + kb;
  const float4 r0 = *(const float4*)(src);
  const float4 r1 = *(const float4*)(src + 4);
  const float4 r2 = *(const float4*)(src + 16);
  const float4 r3 = *(const float4*)(src + 20);
  v16bf v;
  v[0] = (__bf16)r0.x;  v[1] = (__bf16)r0.y;  v[2] = (__bf16)r0.z;  v[3] = (__bf16)r0.w;
  v[4] = (__bf16)r1.x;  v[5] = (__bf16)r1.y;  v[6] = (__bf16)r1.z;  v[7] = (__bf16)r1.w;
  v[8] = (__bf16)r2.x;  v[9] = (__bf16)r2.y;  v[10] = (__bf16)r2.z; v[11] = (__bf16)r2.w;
  v[12] = (__bf16)r3.x; v[13] = (__bf16)r3.y; v[14] = (__bf16)r3.z; v[15] = (__bf16)r3.w;
  *(v16bf*)(Xp + (size_t)(mt * 16 + kt) * FRAG + lane * 16) = v;
}

// w [512,N] f32 -> B-frag-packed bf16 [nt=N/16][kt=16][lane][16] (small, elementwise)
__global__ __launch_bounds__(256) void pack_w_kernel(
    const float* __restrict__ w, __bf16* __restrict__ Wp, int N) {
  const int gn = blockIdx.x * 256 + threadIdx.x;
  const int gk = blockIdx.y;
  const int nt = gn >> 4, n = gn & 15;
  const int kt = gk >> 5, k = gk & 31;
  Wp[((size_t)(nt * 16 + kt)) * FRAG + b_lane(n, k) * 16 + b_idx(k)] =
      (__bf16)w[(size_t)gk * N + gn];
}

// ------------------------------------------------------------ kernel 1: QKV GEMM
// [8192 x 512] x [512 x 1536]; 8 waves/block, 64x256 block tile, 32x64 per wave.
// No LDS; frag streams are L2-resident. Explicit double-buffer so next-kt loads
// overlap current WMMAs (partial s_wait_loadcnt instead of 0x0).
__global__ __launch_bounds__(256) void qkv_gemm_kernel(
    const __bf16* __restrict__ Xp, const __bf16* __restrict__ Wp,
    __bf16* __restrict__ Qp, __bf16* __restrict__ Kp, __bf16* __restrict__ Vp) {
  const int tid = threadIdx.x, wave = tid >> 5, lane = tid & 31;
  const int mw = wave >> 2, nw = wave & 3;
  const int mtBase = blockIdx.y * 4 + mw * 2;
  const int ntBase = blockIdx.x * 16 + nw * 4;

  v8f acc[2][4];
#pragma unroll
  for (int r = 0; r < 2; ++r)
#pragma unroll
    for (int f = 0; f < 4; ++f) acc[r][f] = v8f{};

  const __bf16* aBase0 = Xp + (size_t)(mtBase * 16) * FRAG;
  const __bf16* aBase1 = Xp + (size_t)((mtBase + 1) * 16) * FRAG;
  const __bf16* bBase  = Wp + (size_t)(ntBase * 16) * FRAG;  // f stride 16*FRAG

  v16bf a0c = ld_frag(aBase0);
  v16bf a1c = ld_frag(aBase1);
  v16bf bc[4];
#pragma unroll
  for (int f = 0; f < 4; ++f) bc[f] = ld_frag(bBase + (size_t)(f * 16) * FRAG);

#pragma unroll 2
  for (int kt = 0; kt < 16; ++kt) {
    const int ktn = (kt + 1) & 15;  // last iter reloads kt=0 (discarded)
    __builtin_prefetch(aBase0 + (size_t)((kt + 2) & 15) * FRAG, 0, 3);
    __builtin_prefetch(aBase1 + (size_t)((kt + 2) & 15) * FRAG, 0, 3);
    const v16bf a0n = ld_frag(aBase0 + (size_t)ktn * FRAG);
    const v16bf a1n = ld_frag(aBase1 + (size_t)ktn * FRAG);
    v16bf bn[4];
#pragma unroll
    for (int f = 0; f < 4; ++f) {
      bn[f] = ld_frag(bBase + (size_t)(f * 16 + ktn) * FRAG);
      __builtin_prefetch(bBase + (size_t)(f * 16 + ((kt + 2) & 15)) * FRAG, 0, 3);
    }
#pragma unroll
    for (int f = 0; f < 4; ++f) {
      acc[0][f] = wmma_bf16(a0c, bc[f], acc[0][f]);
      acc[1][f] = wmma_bf16(a1c, bc[f], acc[1][f]);
    }
    a0c = a0n;
    a1c = a1n;
#pragma unroll
    for (int f = 0; f < 4; ++f) bc[f] = bn[f];
  }

  // epilogue: scatter into fragment-packed Q / K / V
  const int ln = lane & 15;
  const int half8 = (lane & 16) ? 8 : 0;
#pragma unroll
  for (int r = 0; r < 2; ++r) {
#pragma unroll
    for (int f = 0; f < 4; ++f) {
#pragma unroll
      for (int e = 0; e < 8; ++e) {
        const int gr = (mtBase + r) * 16 + e + half8;
        const int gc = (ntBase + f) * 16 + ln;
        const int b = gr >> 11, t = gr & (T_ - 1);
        const int sec = gc / D_;
        const int dfull = gc - sec * D_;
        const int h = dfull >> 6, d = dfull & (DH_ - 1);
        const int bh = b * H_ + h;
        const __bf16 val = (__bf16)acc[r][f][e];
        if (sec == 0) {  // Q: A-frag packed [bh][tt=128][dk=2][lane][16]
          const int tt = t >> 4, tm = t & 15, dk = d >> 5, kk = d & 31;
          Qp[((size_t)(bh * 128 + tt) * 2 + dk) * FRAG + a_lane(tm, kk) * 16 +
             a_idx(kk)] = val;
        } else if (sec == 1) {  // K: B-frag packed [bh][jt=128][dk=2][lane][16]
          const int jt = t >> 4, jn = t & 15, dk = d >> 5, kk = d & 31;
          Kp[((size_t)(bh * 128 + jt) * 2 + dk) * FRAG + b_lane(jn, kk) * 16 +
             b_idx(kk)] = val;
        } else {  // V: B-frag packed [bh][jkt=64][dn=4][lane][16]
          const int jkt = t >> 5, vk = t & 31, dn = d >> 4, vn = d & 15;
          Vp[((size_t)(bh * 64 + jkt) * 4 + dn) * FRAG + b_lane(vn, vk) * 16 +
             b_idx(vk)] = val;
        }
      }
    }
  }
}

// ------------------------------------------------------------ kernel 2: attention
// 8 independent waves/block; wave owns 16 query rows; per-wave causal bound.
__global__ __launch_bounds__(256) void attn_kernel(
    const float* __restrict__ msk, const __bf16* __restrict__ Qp,
    const __bf16* __restrict__ Kp, const __bf16* __restrict__ Vp,
    __bf16* __restrict__ Op) {
  __shared__ __align__(32) __bf16 Ps[8 * FRAG];  // per-wave P tile, A-frag layout

  const int tid = threadIdx.x, wave = tid >> 5, lane = tid & 31;
  const int ln = lane & 15;
  const int half8 = (lane & 16) ? 8 : 0;
  const int bh = blockIdx.x;
  const int b = bh >> 3, h = bh & (H_ - 1);
  const int rowBase = blockIdx.y * 128 + wave * 16;
  const int tt = rowBase >> 4;

  const __bf16* qb = Qp + (size_t)(bh * 128 + tt) * 2 * FRAG;
  const v16bf a0 = ld_frag(qb);         // d = 0..31
  const v16bf a1 = ld_frag(qb + FRAG);  // d = 32..63

  v8f o[4] = {v8f{}, v8f{}, v8f{}, v8f{}};
  float mrow[8], lrow[8];
#pragma unroll
  for (int e = 0; e < 8; ++e) { mrow[e] = NEGBIG; lrow[e] = 0.f; }

  __bf16* pw = &Ps[wave * FRAG];

  for (int j0 = 0; j0 < rowBase + 16; j0 += 32) {
    const int jt0 = j0 >> 4;
    const __bf16* kb = Kp + (size_t)(bh * 128 + jt0) * 2 * FRAG;
    const __bf16* vb = Vp + (size_t)(bh * 64 + (j0 >> 5)) * 4 * FRAG;
    __builtin_prefetch(kb + 4 * FRAG, 0, 3);  // next iteration's K tile
    __builtin_prefetch(vb + 4 * FRAG, 0, 3);  // next iteration's V tile

    v8f s0 = v8f{}, s1 = v8f{};
    s0 = wmma_bf16(a0, ld_frag(kb), s0);
    s0 = wmma_bf16(a1, ld_frag(kb + FRAG), s0);
    s1 = wmma_bf16(a0, ld_frag(kb + 2 * FRAG), s1);
    s1 = wmma_bf16(a1, ld_frag(kb + 3 * FRAG), s1);

    const float km0 = msk[(size_t)b * T_ + j0 + ln];
    const float km1 = msk[(size_t)b * T_ + j0 + 16 + ln];

    float scOld[8], p0v[8], p1v[8];
#pragma unroll
    for (int e = 0; e < 8; ++e) {
      const int i = rowBase + e + half8;
      const float e0 = (j0 + ln <= i && km0 != 0.f) ? s0[e] * SCALE_ : NEGBIG;
      const float e1 = (j0 + 16 + ln <= i && km1 != 0.f) ? s1[e] * SCALE_ : NEGBIG;
      const float tm = rmax16(fmaxf(e0, e1));
      const float mn = fmaxf(mrow[e], tm);
      const float p0 = __expf(e0 - mn);
      const float p1 = __expf(e1 - mn);
      const float rs = rsum16(p0 + p1);
      scOld[e] = __expf(mrow[e] - mn);
      mrow[e] = mn;
      lrow[e] = lrow[e] * scOld[e] + rs;
      p0v[e] = p0;
      p1v[e] = p1;
    }

    // P: C-layout regs -> A-frag layout in per-wave LDS (j=ln and j=ln+16)
#pragma unroll
    for (int e = 0; e < 8; ++e) {
      const int lp = (e + half8) + ((ln >= 8) ? 16 : 0);
      pw[lp * 16 + (ln & 7)]     = (__bf16)p0v[e];
      pw[lp * 16 + 8 + (ln & 7)] = (__bf16)p1v[e];
    }
    asm volatile("s_wait_dscnt 0" ::: "memory");
    const v16bf pf = *(const v16bf*)(pw + (size_t)lane * 16);

#pragma unroll
    for (int f = 0; f < 4; ++f) {
#pragma unroll
      for (int e = 0; e < 8; ++e) o[f][e] *= scOld[e];
      o[f] = wmma_bf16(pf, ld_frag(vb + (size_t)f * FRAG), o[f]);
    }
  }

  // epilogue: scatter O into A-frag-packed Op [rt=512][ckt=16][lane][16]
  const int rt = b * 128 + tt;
#pragma unroll
  for (int e = 0; e < 8; ++e) {
    const float inv = 1.0f / lrow[e];
    const int m = e + half8;
#pragma unroll
    for (int f = 0; f < 4; ++f) {
      const int c = h * DH_ + f * 16 + ln;
      const int ckt = c >> 5, kk = c & 31;
      Op[((size_t)rt * 16 + ckt) * FRAG + a_lane(m, kk) * 16 + a_idx(kk)] =
          (__bf16)(o[f][e] * inv);
    }
  }
}

// ------------------------------------------------------------ kernel 3: out-proj
__global__ __launch_bounds__(256) void out_gemm_kernel(
    const float* __restrict__ msk, const __bf16* __restrict__ Op,
    const __bf16* __restrict__ Wp, const float* __restrict__ b_out,
    float* __restrict__ out) {
  const int tid = threadIdx.x, wave = tid >> 5, lane = tid & 31;
  const int mw = wave >> 2, nw = wave & 3;
  const int mtBase = blockIdx.y * 4 + mw * 2;
  const int ntBase = blockIdx.x * 16 + nw * 4;

  v8f acc[2][4];
#pragma unroll
  for (int r = 0; r < 2; ++r)
#pragma unroll
    for (int f = 0; f < 4; ++f) acc[r][f] = v8f{};

  const __bf16* aBase0 = Op + (size_t)(mtBase * 16) * FRAG;
  const __bf16* aBase1 = Op + (size_t)((mtBase + 1) * 16) * FRAG;
  const __bf16* bBase  = Wp + (size_t)(ntBase * 16) * FRAG;

  v16bf a0c = ld_frag(aBase0);
  v16bf a1c = ld_frag(aBase1);
  v16bf bc[4];
#pragma unroll
  for (int f = 0; f < 4; ++f) bc[f] = ld_frag(bBase + (size_t)(f * 16) * FRAG);

#pragma unroll 2
  for (int kt = 0; kt < 16; ++kt) {
    const int ktn = (kt + 1) & 15;
    __builtin_prefetch(aBase0 + (size_t)((kt + 2) & 15) * FRAG, 0, 3);
    __builtin_prefetch(aBase1 + (size_t)((kt + 2) & 15) * FRAG, 0, 3);
    const v16bf a0n = ld_frag(aBase0 + (size_t)ktn * FRAG);
    const v16bf a1n = ld_frag(aBase1 + (size_t)ktn * FRAG);
    v16bf bn[4];
#pragma unroll
    for (int f = 0; f < 4; ++f) {
      bn[f] = ld_frag(bBase + (size_t)(f * 16 + ktn) * FRAG);
      __builtin_prefetch(bBase + (size_t)(f * 16 + ((kt + 2) & 15)) * FRAG, 0, 3);
    }
#pragma unroll
    for (int f = 0; f < 4; ++f) {
      acc[0][f] = wmma_bf16(a0c, bc[f], acc[0][f]);
      acc[1][f] = wmma_bf16(a1c, bc[f], acc[1][f]);
    }
    a0c = a0n;
    a1c = a1n;
#pragma unroll
    for (int f = 0; f < 4; ++f) bc[f] = bn[f];
  }

  const int ln = lane & 15;
  const int half8 = (lane & 16) ? 8 : 0;
#pragma unroll
  for (int r = 0; r < 2; ++r) {
#pragma unroll
    for (int f = 0; f < 4; ++f) {
#pragma unroll
      for (int e = 0; e < 8; ++e) {
        const int gr = (mtBase + r) * 16 + e + half8;
        const int gc = (ntBase + f) * 16 + ln;
        const float v = acc[r][f][e] + b_out[gc];
        out[(size_t)gr * D_ + gc] = v * msk[gr];
      }
    }
  }
}

// ------------------------------------------------------------ launcher

extern "C" void kernel_launch(void* const* d_in, const int* in_sizes, int n_in,
                              void* d_out, int out_size, void* d_ws, size_t ws_size,
                              hipStream_t stream) {
  const float* x     = (const float*)d_in[0];
  const float* m     = (const float*)d_in[1];
  const float* w_qkv = (const float*)d_in[2];
  const float* w_out = (const float*)d_in[3];
  const float* b_out = (const float*)d_in[4];
  float* out = (float*)d_out;

  __bf16* ws = (__bf16*)d_ws;
  const size_t nXp   = (size_t)B_ * T_ * D_;
  const size_t nWqkv = (size_t)D_ * 3 * D_;
  const size_t nWout = (size_t)D_ * D_;
  const size_t nHead = (size_t)B_ * H_ * T_ * DH_;

  __bf16* Xp    = ws;
  __bf16* Wqkvp = Xp + nXp;
  __bf16* Woutp = Wqkvp + nWqkv;
  __bf16* Qp    = Woutp + nWout;
  __bf16* Kp    = Qp + nHead;
  __bf16* Vp    = Kp + nHead;
  __bf16* Op    = Vp + nHead;

  // pack fp32 inputs into WMMA-fragment order (bf16)
  pack_x_kernel<<<dim3((B_ * T_ * D_ / 16) / 256), 256, 0, stream>>>(x, Xp);
  pack_w_kernel<<<dim3((3 * D_) / 256, D_), 256, 0, stream>>>(w_qkv, Wqkvp, 3 * D_);
  pack_w_kernel<<<dim3(D_ / 256, D_), 256, 0, stream>>>(w_out, Woutp, D_);

  // QKV projection
  qkv_gemm_kernel<<<dim3((3 * D_) / 256, (B_ * T_) / 64), 256, 0, stream>>>(
      Xp, Wqkvp, Qp, Kp, Vp);

  // flash attention
  attn_kernel<<<dim3(B_ * H_, T_ / 128), 256, 0, stream>>>(m, Qp, Kp, Vp, Op);

  // output projection
  out_gemm_kernel<<<dim3(D_ / 256, (B_ * T_) / 64), 256, 0, stream>>>(
      m, Op, Woutp, b_out, out);
}
